// GraphEncoder_80058190397889
// MI455X (gfx1250) — matrix operand order repaired and checked by reading, not compiled, not run on hardware
//
#include <hip/hip_runtime.h>
#include <math.h>

// ---------------- problem constants (from setup_inputs) ----------------
#define NODES   20000      // B*Npg = 4*5000
#define CDIM    128
#define HEADS   4
#define HCDIM   512        // H*C
#define EDGES   256000
#define RELS    8
#define LAYERS  2
#define G3      384        // 3*C

typedef __attribute__((ext_vector_type(16))) __bf16 v16bf;
typedef __attribute__((ext_vector_type(8)))  float  v8f;
typedef __attribute__((ext_vector_type(4)))  int    v4i;

__device__ __forceinline__ float sigmoid_f(float x) { return 1.0f / (1.0f + __expf(-x)); }

// 16B global -> LDS copy. Prefer the CDNA5 async path (ASYNCcnt-tracked,
// global_load_async_to_lds_b128); fall back to a synchronous float4 bounce.
// Builtin signature (probe-confirmed by diagnostic): (global v4i* src, lds v4i* dst, imm, imm).
__device__ __forceinline__ void copy16_g2l(const float* g, float* l) {
#if defined(__gfx1250__) && __has_builtin(__builtin_amdgcn_global_load_async_to_lds_b128)
    __builtin_amdgcn_global_load_async_to_lds_b128(
        (__attribute__((address_space(1))) v4i*)(v4i*)(void*)g,
        (__attribute__((address_space(3))) v4i*)(v4i*)(void*)l, 0, 0);
#else
    *(float4*)l = *(const float4*)g;
#endif
}

__device__ __forceinline__ void async_drain() {
#if defined(__gfx1250__) && __has_builtin(__builtin_amdgcn_global_load_async_to_lds_b128)
#if __has_builtin(__builtin_amdgcn_s_wait_asynccnt)
    __builtin_amdgcn_s_wait_asynccnt(0);
#else
    asm volatile("s_wait_asynccnt 0x0" ::: "memory");
#endif
#endif
}

// float atomic max via ordered-int trick (init value must be -inf)
__device__ __forceinline__ void atomicMaxF(float* addr, float v) {
    if (v >= 0.0f) atomicMax((int*)addr, __float_as_int(v));
    else           atomicMin((unsigned int*)addr, __float_as_uint(v));
}

// ---------------- WMMA GEMM: C[M,Ncols] = A[M,128] * B + bias ----------------
// BT==false: B is [128, Ncols] row-major (B[k*ldb + n])   -> x @ W_gat
// BT==true : B is [Ncols, 128] row-major (B[n*ldb + k])   -> x @ W^T (GRU)
// block = (32,4): 4 waves; one 16-row m-tile, 4 adjacent 16-col n-tiles.
// A tile (16x128) and B tile (64 cols x 128 K) are staged into LDS once per
// block via async global->LDS (ASYNCcnt), then 4 fully unrolled
// v_wmma_f32_16x16x32_bf16 steps consume them.
// Fragment layouts follow CDNA5 ISA 7.12.2 (wave32).
template<bool BT>
__global__ __launch_bounds__(128) void wmma_gemm128(
        const float* __restrict__ A, const float* __restrict__ B,
        const float* __restrict__ bias, float* __restrict__ C,
        int Ncols, int ldb) {
    constexpr int ASTR = 132;                       // A row stride (pad, 16B-mult)
    constexpr int BSTR = BT ? 132 : 68;             // B tile stride (pad, 16B-mult)
    constexpr int BS_ELEMS = BT ? 64 * 132 : 128 * 68;
    __shared__ float As[16 * ASTR];
    __shared__ float Bs[BS_ELEMS];

    const int tid    = threadIdx.y * 32 + threadIdx.x;   // 0..127
    const int mbase  = blockIdx.y * 16;
    const int nbase0 = blockIdx.x * 64;

    // ---- stage A: 16 rows x 128 K  (512 x b128 chunks) ----
#pragma unroll
    for (int i = 0; i < 4; ++i) {
        int t  = tid + i * 128;            // 0..511
        int m  = t >> 5;                   // row in tile
        int kf = (t & 31) * 4;             // float4-aligned K offset
        copy16_g2l(A + (size_t)(mbase + m) * CDIM + kf, As + m * ASTR + kf);
    }
    // ---- stage B: 64 cols x 128 K  (2048 x b128 chunks) ----
    if constexpr (BT) {
#pragma unroll
        for (int i = 0; i < 16; ++i) {
            int t  = tid + i * 128;        // 0..2047
            int n  = t >> 5;
            int kf = (t & 31) * 4;
            copy16_g2l(B + (size_t)(nbase0 + n) * ldb + kf, Bs + n * BSTR + kf);
        }
    } else {
#pragma unroll
        for (int i = 0; i < 16; ++i) {
            int t  = tid + i * 128;
            int k  = t >> 4;
            int nf = (t & 15) * 4;
            copy16_g2l(B + (size_t)k * ldb + nbase0 + nf, Bs + k * BSTR + nf);
        }
    }
    async_drain();
    __syncthreads();

    const int lane = threadIdx.x;
    const int l16  = lane & 15;
    const int kgrp = lane >> 4;                       // 0/1 half-wave
    const int nloc = threadIdx.y * 16 + l16;          // tile-local column
    const float* Arow = As + l16 * ASTR;

    v8f acc = {};
#pragma unroll
    for (int kb = 0; kb < CDIM; kb += 32) {
        v16bf afrag, bfrag;
        // A 16x32 bf16 fragment: lanes 0-15 rows M; VGPR0-3 K=0..7, VGPR4-7 K=16..23 (+8 if kgrp)
#pragma unroll
        for (int v = 0; v < 8; ++v) {
            int K = kb + v * 2 + (v >= 4 ? 8 : 0) + kgrp * 8;
            afrag[2 * v]     = (__bf16)Arow[K];
            afrag[2 * v + 1] = (__bf16)Arow[K + 1];
        }
        // B 32x16 bf16 fragment: lane holds column n = lane%16, K = kgrp*8 + i (+8 if i >= 8)
#pragma unroll
        for (int i = 0; i < 16; ++i) {
            int K = kb + kgrp * 8 + i + (i >= 8 ? 8 : 0);
            float bv = BT ? Bs[nloc * BSTR + K] : Bs[K * BSTR + nloc];
            bfrag[i] = (__bf16)bv;
        }
        acc = __builtin_amdgcn_wmma_f32_16x16x32_bf16(false, afrag, false, bfrag,
                                                      (short)0, acc, false, false);
    }
    // D 16x16 f32: lane = column n, VGPR r -> row r + 8*kgrp
    const int   ncol = nbase0 + nloc;
    const float bv   = bias ? bias[ncol] : 0.0f;
#pragma unroll
    for (int r = 0; r < 8; ++r) {
        int m = mbase + r + kgrp * 8;
        C[(size_t)m * Ncols + ncol] = acc[r] + bv;
    }
}

// ---------------- attention scores: s[n,r,h] = sum_c xt[n,h,c]*a[r,h,c] ----------------
__global__ void attn_scores_k(const float* __restrict__ xt, const float* __restrict__ a_s,
                              const float* __restrict__ a_d, float* __restrict__ ssrc,
                              float* __restrict__ sdst) {
    int idx = blockIdx.x * blockDim.x + threadIdx.x;   // n*32 + r*4 + h
    if (idx >= NODES * 32) return;
    int n = idx >> 5, rh = idx & 31, r = rh >> 2, h = rh & 3;
    const float* xrow = xt + (size_t)n * HCDIM + h * CDIM;
    const float* as   = a_s + (size_t)(r * HEADS + h) * CDIM;
    const float* ad   = a_d + (size_t)(r * HEADS + h) * CDIM;
    float ss = 0.0f, sd = 0.0f;
#pragma unroll 4
    for (int c = 0; c < CDIM; ++c) { float xv = xrow[c]; ss += xv * as[c]; sd += xv * ad[c]; }
    ssrc[idx] = ss;
    sdst[idx] = sd;
}

__global__ void fill_k(float* __restrict__ p, float v, int n) {
    int i = blockIdx.x * blockDim.x + threadIdx.x;
    if (i < n) p[i] = v;
}

// ---------------- edge logits + leaky relu + segment max ----------------
__global__ void edge_logits_k(const float* __restrict__ ssrc, const float* __restrict__ sdst,
                              const int* __restrict__ src, const int* __restrict__ dst,
                              const int* __restrict__ et, float* __restrict__ lg,
                              float* __restrict__ mbuf) {
    int e = blockIdx.x * blockDim.x + threadIdx.x;
    if (e >= EDGES) return;
    int s = src[e], d = dst[e], r = et[e];
    const float4 a = *(const float4*)(ssrc + (size_t)s * 32 + r * 4);
    const float4 b = *(const float4*)(sdst + (size_t)d * 32 + r * 4);
    float l[4] = {a.x + b.x, a.y + b.y, a.z + b.z, a.w + b.w};
#pragma unroll
    for (int h = 0; h < 4; ++h) {
        float v = l[h];
        v = v > 0.0f ? v : 0.2f * v;        // LeakyReLU(0.2)
        l[h] = v;
        atomicMaxF(mbuf + (size_t)d * 4 + h, v);
    }
    *(float4*)(lg + (size_t)e * 4) = make_float4(l[0], l[1], l[2], l[3]);
}

// nodes with no incoming edges: m stays -inf -> 0 (matches jnp.where(isfinite))
__global__ void fix_m_k(float* __restrict__ mbuf) {
    int i = blockIdx.x * blockDim.x + threadIdx.x;
    if (i >= NODES * 4) return;
    float v = mbuf[i];
    if (v < -1e30f) mbuf[i] = 0.0f;
}

// ---------------- exp(logit - m[dst]) + segment sum (in-place over lg) ----------------
__global__ void edge_exp_k(float* __restrict__ lg, const float* __restrict__ mbuf,
                           const int* __restrict__ dst, float* __restrict__ den) {
    int e = blockIdx.x * blockDim.x + threadIdx.x;
    if (e >= EDGES) return;
    int d = dst[e];
    float4 l  = *(const float4*)(lg + (size_t)e * 4);
    float4 mm = *(const float4*)(mbuf + (size_t)d * 4);
    float ex[4] = {__expf(l.x - mm.x), __expf(l.y - mm.y), __expf(l.z - mm.z), __expf(l.w - mm.w)};
    *(float4*)(lg + (size_t)e * 4) = make_float4(ex[0], ex[1], ex[2], ex[3]);
#pragma unroll
    for (int h = 0; h < 4; ++h) atomicAdd(den + (size_t)d * 4 + h, ex[h]);
}

// ---------------- aggregation: msg[dst,c] += (1/H) * sum_h alpha[e,h]*xt[src,h,c] ----------------
// one wave per edge; xt (41MB) is L2-resident so gathers are L2-bandwidth bound.
__global__ void edge_agg_k(const float* __restrict__ xt, const float* __restrict__ exb,
                           const float* __restrict__ den, const int* __restrict__ src,
                           const int* __restrict__ dst, float* __restrict__ msg) {
    int gtid = blockIdx.x * blockDim.x + threadIdx.x;
    int e    = gtid >> 5;
    int lane = threadIdx.x & 31;
    if (e >= EDGES) return;
    int s = src[e], d = dst[e];
    int c = lane * 4;
    __builtin_prefetch(msg + (size_t)d * CDIM + c, 1, 0);   // prefetch-for-write of scatter dest
    const float* exr = exb + (size_t)e * 4;
    const float* dnr = den + (size_t)d * 4;
    float al[4];
#pragma unroll
    for (int h = 0; h < 4; ++h) al[h] = exr[h] / (dnr[h] + 1e-16f) * 0.25f;   // fold head-mean
    float4 acc = make_float4(0.f, 0.f, 0.f, 0.f);
#pragma unroll
    for (int h = 0; h < 4; ++h) {
        float4 v = *(const float4*)(xt + (size_t)s * HCDIM + h * CDIM + c);
        acc.x += al[h] * v.x; acc.y += al[h] * v.y; acc.z += al[h] * v.z; acc.w += al[h] * v.w;
    }
    float* mp = msg + (size_t)d * CDIM + c;
    atomicAdd(mp + 0, acc.x); atomicAdd(mp + 1, acc.y);
    atomicAdd(mp + 2, acc.z); atomicAdd(mp + 3, acc.w);
}

__global__ void add_bias_k(float* __restrict__ msg, const float* __restrict__ bias) {
    int i = blockIdx.x * blockDim.x + threadIdx.x;
    if (i >= NODES * CDIM) return;
    msg[i] += bias[i & (CDIM - 1)];
}

// ---------------- GRU gate elementwise (in-place update of x) ----------------
__global__ void gru_gates_k(const float* __restrict__ gi, const float* __restrict__ gh,
                            float* __restrict__ x) {
    int idx = blockIdx.x * blockDim.x + threadIdx.x;
    if (idx >= NODES * CDIM) return;
    int n = idx >> 7, c = idx & 127;
    size_t b = (size_t)n * G3 + c;
    float ir = gi[b], iz = gi[b + CDIM], in = gi[b + 2 * CDIM];
    float hr = gh[b], hz = gh[b + CDIM], hn = gh[b + 2 * CDIM];
    float r = sigmoid_f(ir + hr);
    float z = sigmoid_f(iz + hz);
    float nn = tanhf(in + r * hn);
    x[idx] = (1.0f - z) * nn + z * x[idx];
}

// ---------------- final LayerNorm over C (one wave per node) ----------------
__global__ void layernorm_k(const float* __restrict__ x, const float* __restrict__ g,
                            const float* __restrict__ b, float* __restrict__ out) {
    int gtid = blockIdx.x * blockDim.x + threadIdx.x;
    int n    = gtid >> 5;
    int lane = threadIdx.x & 31;
    if (n >= NODES) return;
    int c = lane * 4;
    float4 v = *(const float4*)(x + (size_t)n * CDIM + c);
    float s  = v.x + v.y + v.z + v.w;
    float sq = v.x * v.x + v.y * v.y + v.z * v.z + v.w * v.w;
#pragma unroll
    for (int off = 16; off > 0; off >>= 1) {
        s  += __shfl_xor(s,  off, 32);
        sq += __shfl_xor(sq, off, 32);
    }
    float mu  = s * (1.0f / CDIM);
    float var = sq * (1.0f / CDIM) - mu * mu;
    float inv = rsqrtf(var + 1e-5f);
    float4 o;
    o.x = (v.x - mu) * inv * g[c + 0] + b[c + 0];
    o.y = (v.y - mu) * inv * g[c + 1] + b[c + 1];
    o.z = (v.z - mu) * inv * g[c + 2] + b[c + 2];
    o.w = (v.w - mu) * inv * g[c + 3] + b[c + 3];
    *(float4*)(out + (size_t)n * CDIM + c) = o;
}

__global__ void copy_k(const float* __restrict__ src, float* __restrict__ dst, int n) {
    int i = blockIdx.x * blockDim.x + threadIdx.x;
    if (i < n) dst[i] = src[i];
}

// ---------------- orchestration ----------------
extern "C" void kernel_launch(void* const* d_in, const int* in_sizes, int n_in,
                              void* d_out, int out_size, void* d_ws, size_t ws_size,
                              hipStream_t stream) {
    const float* emb     = (const float*)d_in[0];
    const int*   eidx    = (const int*)  d_in[1];
    const int*   etype   = (const int*)  d_in[2];
    const float* W_gat   = (const float*)d_in[3];
    const float* att_src = (const float*)d_in[4];
    const float* att_dst = (const float*)d_in[5];
    const float* b_gat   = (const float*)d_in[6];
    const float* W_ih    = (const float*)d_in[7];
    const float* W_hh    = (const float*)d_in[8];
    const float* b_ih    = (const float*)d_in[9];
    const float* b_hh    = (const float*)d_in[10];
    const float* ln_g    = (const float*)d_in[11];
    const float* ln_b    = (const float*)d_in[12];
    float*       out     = (float*)d_out;

    const int* src = eidx;
    const int* dst = eidx + EDGES;

    // workspace carve-up (floats), ~133 MB total
    float* ws   = (float*)d_ws;
    float* x    = ws;                    ws += (size_t)NODES * CDIM;    // state
    float* xt   = ws;                    ws += (size_t)NODES * HCDIM;   // x @ W_gat  (L2-resident, 41MB)
    float* ssrc = ws;                    ws += (size_t)NODES * 32;
    float* sdst = ws;                    ws += (size_t)NODES * 32;
    float* mbuf = ws;                    ws += (size_t)NODES * 4;
    float* den  = ws;                    ws += (size_t)NODES * 4;
    float* lg   = ws;                    ws += (size_t)EDGES * 4;       // logits -> exp (in place)
    float* msg  = ws;                    ws += (size_t)NODES * CDIM;
    float* gi   = ws;                    ws += (size_t)NODES * G3;
    float* gh   = ws;                    ws += (size_t)NODES * G3;

    const dim3 blk256(256);
    const dim3 gemmBlk(32, 4);
    const int NC  = NODES * CDIM;                        // 2.56M
    const int eBlocks   = (EDGES + 255) / 256;
    const int ncBlocks  = (NC + 255) / 256;
    const int n32Blocks = (NODES * 32 + 255) / 256;
    const int n4Blocks  = (NODES * 4 + 255) / 256;
    const int aggBlocks = (EDGES * 32 + 255) / 256;      // wave per edge
    const int lnBlocks  = (NODES * 32 + 255) / 256;      // wave per node

    copy_k<<<ncBlocks, blk256, 0, stream>>>(emb, x, NC);

    for (int l = 0; l < LAYERS; ++l) {
        const float* Wg = W_gat   + (size_t)l * CDIM * HCDIM;
        const float* as = att_src + (size_t)l * RELS * HEADS * CDIM;
        const float* ad = att_dst + (size_t)l * RELS * HEADS * CDIM;
        const float* bg = b_gat   + (size_t)l * CDIM;

        // xt = x @ W_gat[l]   [N,128] x [128,512]  (bf16 WMMA, f32 accum)
        wmma_gemm128<false><<<dim3(HCDIM / 64, NODES / 16), gemmBlk, 0, stream>>>(
            x, Wg, nullptr, xt, HCDIM, HCDIM);

        attn_scores_k<<<n32Blocks, blk256, 0, stream>>>(xt, as, ad, ssrc, sdst);

        fill_k<<<n4Blocks, blk256, 0, stream>>>(mbuf, -INFINITY, NODES * 4);
        fill_k<<<n4Blocks, blk256, 0, stream>>>(den, 0.0f, NODES * 4);
        fill_k<<<ncBlocks, blk256, 0, stream>>>(msg, 0.0f, NC);

        edge_logits_k<<<eBlocks, blk256, 0, stream>>>(ssrc, sdst, src, dst, etype, lg, mbuf);
        fix_m_k<<<n4Blocks, blk256, 0, stream>>>(mbuf);
        edge_exp_k<<<eBlocks, blk256, 0, stream>>>(lg, mbuf, dst, den);
        edge_agg_k<<<aggBlocks, blk256, 0, stream>>>(xt, lg, den, src, dst, msg);
        add_bias_k<<<ncBlocks, blk256, 0, stream>>>(msg, bg);

        // GRU: gi = msg @ W_ih^T + b_ih ; gh = x @ W_hh^T + b_hh   [N,128] x [128,384]
        wmma_gemm128<true><<<dim3(G3 / 64, NODES / 16), gemmBlk, 0, stream>>>(
            msg, W_ih, b_ih, gi, G3, CDIM);
        wmma_gemm128<true><<<dim3(G3 / 64, NODES / 16), gemmBlk, 0, stream>>>(
            x, W_hh, b_hh, gh, G3, CDIM);
        gru_gates_k<<<ncBlocks, blk256, 0, stream>>>(gi, gh, x);
    }

    layernorm_k<<<lnBlocks, blk256, 0, stream>>>(x, ln_g, ln_b, out);
}
